// MultiHeadAttention_9912784519895
// MI455X (gfx1250) — matrix operand order repaired and checked by reading, not compiled, not run on hardware
//
#include <hip/hip_runtime.h>
#include <hip/hip_bf16.h>

// ---------------------------------------------------------------------------
// entmax-1.5 MHA for MI455X (gfx1250, wave32)
//   - all GEMMs: v_wmma_f32_16x16x32_f16, f16-resident operands
//   - staging:   global_load_async_to_lds_b128 (ASYNCcnt) + double-buffered LDS
//                branch-free main loop (last K-tile peeled)
//   - entmax15:  register-resident bisection (replaces reference's full sort)
//   - streams:   non-temporal loads/stores for the O(S^2) attn tensors
//   - epilogue:  fused residual + LayerNorm
// ---------------------------------------------------------------------------

typedef __attribute__((ext_vector_type(16))) _Float16 v16h;
typedef __attribute__((ext_vector_type(8)))  _Float16 v8h;
typedef __attribute__((ext_vector_type(4)))  _Float16 v4h;
typedef __attribute__((ext_vector_type(8)))  float    v8f;

#define S_SEQ  3072
#define DMODEL 512
#define NHEAD  8
#define DHEAD  64

#define BM 128
#define BN 64
#define BK 32
#define ASTR 40   // LDS row stride in halfs (80 B, keeps 16B alignment, skews banks)
#define BSTR 40

// ---------------------------------------------------------------------------
// CDNA5 async copy: 16 bytes global -> LDS, tracked by ASYNCcnt.
// ---------------------------------------------------------------------------
__device__ __forceinline__ void async_copy_b128(unsigned lds_off, unsigned goff,
                                                unsigned long long base) {
    asm volatile("global_load_async_to_lds_b128 %0, %1, %2"
                 :: "v"(lds_off), "v"(goff), "s"(base) : "memory");
}
__device__ __forceinline__ void wait_async_le3() {
    asm volatile("s_wait_asynccnt 0x3" ::: "memory");
}
__device__ __forceinline__ void wait_async_0() {
    asm volatile("s_wait_asynccnt 0x0" ::: "memory");
}
// Low 32 bits of the flat address of a shared pointer == LDS byte offset.
__device__ __forceinline__ unsigned lds_off_of(const void* p) {
    return (unsigned)(unsigned long long)p;
}

// ---------------------------------------------------------------------------
// f16 WMMA GEMM:  C[z] = (A[z] * B[z]^T-layout + bias) * out_scale
//   A (f16): element (m,k) = A[m*lda + k]
//   B (f16): element (k,n) = B[n*ldb + k]     (always K-contiguous "BT" layout)
//   C: f32 or f16, normal (C[m*ldc+n]) or transposed (C[n*ldc+m]); optional NT
// grid = (M/128, N/64, batch), block = 256 threads = 8 waves (wave tile 32x32)
// ---------------------------------------------------------------------------
template <bool OUT_HALF, bool OUT_TRANS, bool OUT_NT>
__global__ __launch_bounds__(256) void gemm_h_wmma(
    const _Float16* __restrict__ A, long long lda, long long azs,
    const _Float16* __restrict__ B, long long ldb, long long bzs,
    const float* __restrict__ bias,
    void* __restrict__ Cv, long long ldc, long long czs,
    int K, float out_scale)
{
    __shared__ __align__(16) _Float16 As[2][BM][ASTR];
    __shared__ __align__(16) _Float16 Bs[2][BN][BSTR];

    A += (long long)blockIdx.z * azs;
    B += (long long)blockIdx.z * bzs;

    const int m0   = blockIdx.x * BM;
    const int n0   = blockIdx.y * BN;
    const int tid  = threadIdx.x;
    const int lane = tid & 31;
    const int w    = tid >> 5;
    const int wm   = (w & 3) * 32;
    const int wn   = (w >> 2) * 32;

    // ---- per-thread async-staging addresses (32-bit offsets, computed once) --
    const int srow = tid >> 2;             // 0..63
    const unsigned c16 = (tid & 3) * 16u;  // byte offset of 8-half chunk in row
    const unsigned aoff0 = (unsigned)((m0 + srow)      * lda) * 2u + c16;
    const unsigned aoff1 = (unsigned)((m0 + 64 + srow) * lda) * 2u + c16;
    const unsigned boff  = (unsigned)((n0 + srow)      * ldb) * 2u + c16;
    const unsigned lds_a0 = lds_off_of(&As[0][srow][(tid & 3) * 8]);
    const unsigned lds_a1 = lds_off_of(&As[0][64 + srow][(tid & 3) * 8]);
    const unsigned lds_b  = lds_off_of(&Bs[0][srow][(tid & 3) * 8]);
    const unsigned ASZ = BM * ASTR * 2u;
    const unsigned BSZ = BN * BSTR * 2u;
    const unsigned long long abase = (unsigned long long)A;
    const unsigned long long bbase = (unsigned long long)B;

    const v8f vzero = {0.f, 0.f, 0.f, 0.f, 0.f, 0.f, 0.f, 0.f};
    v8f acc[2][2] = {{vzero, vzero}, {vzero, vzero}};

    // fragment addressing per CDNA5 wave32 WMMA layouts
    const int r    = lane & 15;
    const int hsel = lane >> 4;
    const int ab   = hsel * 8;    // A: halves0-7 = K[ab..], 8-15 = K[16+ab..]
    const int kb2  = hsel * 16;   // B: same K order as A (halves0-7 = K[ab..], 8-15 = K[16+ab..])
    union Frag { v16h v; v8h h[2]; };

    auto compute_tile = [&](int bsel) {
        Frag a[2], b[2];
        #pragma unroll
        for (int i = 0; i < 2; ++i) {
            int row = wm + i * 16 + r;
            a[i].h[0] = *(const v8h*)&As[bsel][row][ab];
            a[i].h[1] = *(const v8h*)&As[bsel][row][16 + ab];
        }
        #pragma unroll
        for (int j = 0; j < 2; ++j) {
            int col = wn + j * 16 + r;
            b[j].h[0] = *(const v8h*)&Bs[bsel][col][ab];
            b[j].h[1] = *(const v8h*)&Bs[bsel][col][16 + ab];
        }
        #pragma unroll
        for (int i = 0; i < 2; ++i)
            #pragma unroll
            for (int j = 0; j < 2; ++j)
                acc[i][j] = __builtin_amdgcn_wmma_f32_16x16x32_f16(
                    false, a[i].v, false, b[j].v, (short)0, acc[i][j], false, false);
    };

    // stage tile 0 into buffer 0 (3 async ops per thread per tile)
    async_copy_b128(lds_a0, aoff0, abase);
    async_copy_b128(lds_a1, aoff1, abase);
    async_copy_b128(lds_b,  boff,  bbase);

    // branch-free main loop: stage tile t+1 while computing tile t
    int buf = 0;
    for (int k0 = 0; k0 + BK < K; k0 += BK) {
        const unsigned kb = (unsigned)(k0 + BK) * 2u;
        const unsigned da = (buf ^ 1) ? ASZ : 0u;
        const unsigned db = (buf ^ 1) ? BSZ : 0u;
        async_copy_b128(lds_a0 + da, aoff0 + kb, abase);
        async_copy_b128(lds_a1 + da, aoff1 + kb, abase);
        async_copy_b128(lds_b  + db, boff  + kb, bbase);
        wait_async_le3();   // current tile's 3 ops retired (per-wave, in order)
        __syncthreads();
        compute_tile(buf);
        __syncthreads();    // all LDS reads of `buf` done before it is re-staged
        buf ^= 1;
    }
    // peeled last tile
    wait_async_0();
    __syncthreads();
    compute_tile(buf);

    // ---- epilogue: elem e -> M = e + 8*hsel, N = lane%16 ----
    #pragma unroll
    for (int i = 0; i < 2; ++i) {
        #pragma unroll
        for (int j = 0; j < 2; ++j) {
            const int col = n0 + wn + j * 16 + r;
            const float bv = bias ? bias[col] : 0.0f;
            #pragma unroll
            for (int e = 0; e < 8; ++e) {
                const int rowm = m0 + wm + i * 16 + e + 8 * hsel;
                const float val = (acc[i][j][e] + bv) * out_scale;
                const long long cidx = OUT_TRANS
                    ? (long long)col * ldc + rowm + (long long)blockIdx.z * czs
                    : (long long)rowm * ldc + col + (long long)blockIdx.z * czs;
                if (OUT_HALF) {
                    _Float16 hv = (_Float16)val;
                    if (OUT_NT) __builtin_nontemporal_store(hv, (_Float16*)Cv + cidx);
                    else        ((_Float16*)Cv)[cidx] = hv;
                } else {
                    if (OUT_NT) __builtin_nontemporal_store(val, (float*)Cv + cidx);
                    else        ((float*)Cv)[cidx] = val;
                }
            }
        }
    }
}

// ---------------------------------------------------------------------------
// Block reductions (256 threads = 8 waves of 32)
// ---------------------------------------------------------------------------
__device__ __forceinline__ float block_reduce_sum(float v, float* sh) {
    for (int off = 16; off > 0; off >>= 1) v += __shfl_down(v, off, 32);
    int lane = threadIdx.x & 31, w = threadIdx.x >> 5;
    if (lane == 0) sh[w] = v;
    __syncthreads();
    if (w == 0) {
        float t = (lane < 8) ? sh[lane] : 0.0f;
        for (int off = 4; off > 0; off >>= 1) t += __shfl_down(t, off, 32);
        if (lane == 0) sh[0] = t;
    }
    __syncthreads();
    float r = sh[0];
    __syncthreads();
    return r;
}

__device__ __forceinline__ float block_reduce_max(float v, float* sh) {
    for (int off = 16; off > 0; off >>= 1) v = fmaxf(v, __shfl_down(v, off, 32));
    int lane = threadIdx.x & 31, w = threadIdx.x >> 5;
    if (lane == 0) sh[w] = v;
    __syncthreads();
    if (w == 0) {
        float t = (lane < 8) ? sh[lane] : -3.0e38f;
        for (int off = 4; off > 0; off >>= 1) t = fmaxf(t, __shfl_down(t, off, 32));
        if (lane == 0) sh[0] = t;
    }
    __syncthreads();
    float r = sh[0];
    __syncthreads();
    return r;
}

// ---------------------------------------------------------------------------
// entmax-1.5 on one row of 3072 logits (in place, register-resident).
// Streams: NT load of logits, NT stores of f32 + f16 probabilities.
// ---------------------------------------------------------------------------
#define ENT_TPB 256
#define ENT_PER (S_SEQ / ENT_TPB)   // 12

__global__ __launch_bounds__(ENT_TPB) void entmax15_kernel(
    float* __restrict__ attn, _Float16* __restrict__ attnh)
{
    __shared__ float sh[9];
    float*    p  = attn  + (long long)blockIdx.x * S_SEQ;
    _Float16* ph = attnh + (long long)blockIdx.x * S_SEQ;

    float x[ENT_PER];
    float mx = -3.0e38f;
    #pragma unroll
    for (int i = 0; i < ENT_PER; ++i) {
        x[i] = __builtin_nontemporal_load(p + threadIdx.x + i * ENT_TPB);
        mx = fmaxf(mx, x[i]);
    }
    mx = block_reduce_max(mx, sh);
    #pragma unroll
    for (int i = 0; i < ENT_PER; ++i) x[i] = (x[i] - mx) * 0.5f;

    float lo = -1.0f, hi = 0.0f;   // after shift, max(X)=0 => tau* in [-1,0]
    for (int it = 0; it < 30; ++it) {
        float tau = 0.5f * (lo + hi);
        float s = 0.0f;
        #pragma unroll
        for (int i = 0; i < ENT_PER; ++i) {
            float d = fmaxf(x[i] - tau, 0.0f);
            s += d * d;
        }
        s = block_reduce_sum(s, sh);
        if (s >= 1.0f) lo = tau; else hi = tau;
    }
    float tau = 0.5f * (lo + hi);
    #pragma unroll
    for (int i = 0; i < ENT_PER; ++i) {
        float d = fmaxf(x[i] - tau, 0.0f);
        float pv = d * d;
        __builtin_nontemporal_store(pv, p + threadIdx.x + i * ENT_TPB);
        __builtin_nontemporal_store((_Float16)pv, ph + threadIdx.x + i * ENT_TPB);
    }
}

// ---------------------------------------------------------------------------
// out = LayerNorm(x + y) * gamma + beta   (one row of 512 per block)
// ---------------------------------------------------------------------------
__global__ __launch_bounds__(256) void residual_ln_kernel(
    const float* __restrict__ x, const float* __restrict__ y,
    const float* __restrict__ gamma, const float* __restrict__ beta,
    float* __restrict__ out)
{
    __shared__ float sh[9];
    const long long row = blockIdx.x;
    float v[2];
    #pragma unroll
    for (int j = 0; j < 2; ++j) {
        int c = threadIdx.x + j * 256;
        v[j] = x[row * DMODEL + c] + y[row * DMODEL + c];
    }
    float mu = block_reduce_sum(v[0] + v[1], sh) * (1.0f / DMODEL);
    float d0 = v[0] - mu, d1 = v[1] - mu;
    float var = block_reduce_sum(d0 * d0 + d1 * d1, sh) * (1.0f / DMODEL);
    float rstd = rsqrtf(var + 1e-6f);
    #pragma unroll
    for (int j = 0; j < 2; ++j) {
        int c = threadIdx.x + j * 256;
        out[row * DMODEL + c] = (v[j] - mu) * rstd * gamma[c] + beta[c];
    }
}

// ---------------------------------------------------------------------------
// f32 -> f16 converters (plain + 512x512 transpose for weights)
// ---------------------------------------------------------------------------
__global__ __launch_bounds__(256) void cvt_f16_kernel(
    const float* __restrict__ in, _Float16* __restrict__ out, long long n4)
{
    long long i = (long long)blockIdx.x * 256 + threadIdx.x;
    if (i >= n4) return;
    float4 v = ((const float4*)in)[i];
    v4h o = {(_Float16)v.x, (_Float16)v.y, (_Float16)v.z, (_Float16)v.w};
    ((v4h*)out)[i] = o;
}

__global__ __launch_bounds__(256) void cvt_transpose_kernel(
    const float* __restrict__ W, _Float16* __restrict__ Wt)  // Wt[n][k] = W[k][n]
{
    __shared__ _Float16 t[32][33];
    const int bn = blockIdx.x * 32, bk = blockIdx.y * 32;
    const int tx = threadIdx.x & 31, ty = threadIdx.x >> 5;   // 32 x 8
    #pragma unroll
    for (int i = 0; i < 32; i += 8)
        t[ty + i][tx] = (_Float16)W[(long long)(bk + ty + i) * DMODEL + bn + tx];
    __syncthreads();
    #pragma unroll
    for (int i = 0; i < 32; i += 8)
        Wt[(long long)(bn + ty + i) * DMODEL + bk + tx] = t[tx][ty + i];
}

// ---------------------------------------------------------------------------
extern "C" void kernel_launch(void* const* d_in, const int* in_sizes, int n_in,
                              void* d_out, int out_size, void* d_ws, size_t ws_size,
                              hipStream_t stream)
{
    const float* x     = (const float*)d_in[0];
    const float* Wq    = (const float*)d_in[1];
    const float* bq    = (const float*)d_in[2];
    const float* Wk    = (const float*)d_in[3];
    const float* bk    = (const float*)d_in[4];
    const float* Wv    = (const float*)d_in[5];
    const float* bv    = (const float*)d_in[6];
    const float* Wo    = (const float*)d_in[7];
    const float* bo    = (const float*)d_in[8];
    const float* gamma = (const float*)d_in[9];
    const float* beta  = (const float*)d_in[10];

    float* out_main = (float*)d_out;                               // [S, D]
    float* attn     = (float*)d_out + (long long)S_SEQ * DMODEL;   // [H, S, S] f32

    const long long SD = (long long)S_SEQ * DMODEL;
    const long long SS = (long long)S_SEQ * S_SEQ;
    const long long WW = (long long)DMODEL * DMODEL;

    char* wp = (char*)d_ws;
    _Float16* xh    = (_Float16*)wp; wp += SD * 2;        // x   f16 [S, D]
    _Float16* qh    = (_Float16*)wp; wp += SD * 2;        // q/8 f16 [S, H*DK]
    _Float16* kh    = (_Float16*)wp; wp += SD * 2;        // k   f16 [S, H*DK]
    _Float16* vt    = (_Float16*)wp; wp += SD * 2;        // v^T f16 [H*DV, S]
    _Float16* abuf  = (_Float16*)wp; wp += SD * 2;        // attn@v f16 [S, H*DV]
    _Float16* WqT   = (_Float16*)wp; wp += WW * 2;
    _Float16* WkT   = (_Float16*)wp; wp += WW * 2;
    _Float16* WvT   = (_Float16*)wp; wp += WW * 2;
    _Float16* WoT   = (_Float16*)wp; wp += WW * 2;
    float*    ybuf  = (float*)wp;    wp += SD * 4;        // out-proj f32 [S, D]
    _Float16* attnh = (_Float16*)wp;                      // attn f16 [H, S, S]

    dim3 blk(256);
    dim3 tgrid(DMODEL / 32, DMODEL / 32);

    // precision prep: f16 copies; weights transposed so B is K-contiguous
    cvt_f16_kernel<<<dim3((unsigned)((SD / 4 + 255) / 256)), blk, 0, stream>>>(x, xh, SD / 4);
    cvt_transpose_kernel<<<tgrid, blk, 0, stream>>>(Wq, WqT);
    cvt_transpose_kernel<<<tgrid, blk, 0, stream>>>(Wk, WkT);
    cvt_transpose_kernel<<<tgrid, blk, 0, stream>>>(Wv, WvT);
    cvt_transpose_kernel<<<tgrid, blk, 0, stream>>>(Wo, WoT);

    // QKV projections (q pre-scaled by 1/sqrt(dk)); v written transposed
    gemm_h_wmma<true, false, false><<<dim3(S_SEQ / BM, DMODEL / BN, 1), blk, 0, stream>>>(
        xh, DMODEL, 0, WqT, DMODEL, 0, bq, qh, DMODEL, 0, DMODEL, 0.125f);
    gemm_h_wmma<true, false, false><<<dim3(S_SEQ / BM, DMODEL / BN, 1), blk, 0, stream>>>(
        xh, DMODEL, 0, WkT, DMODEL, 0, bk, kh, DMODEL, 0, DMODEL, 1.0f);
    gemm_h_wmma<true, true, false><<<dim3(S_SEQ / BM, DMODEL / BN, 1), blk, 0, stream>>>(
        xh, DMODEL, 0, WvT, DMODEL, 0, bv, vt, S_SEQ, 0, DMODEL, 1.0f);

    // logits[h] = (q_h/8) @ k_h^T -> f32 attn region of d_out (streaming, NT)
    gemm_h_wmma<false, false, true><<<dim3(S_SEQ / BM, S_SEQ / BN, NHEAD), blk, 0, stream>>>(
        qh, DMODEL, DHEAD, kh, DMODEL, DHEAD, nullptr,
        attn, S_SEQ, SS, DHEAD, 1.0f);

    // entmax-1.5 in place; also emits f16 probabilities
    entmax15_kernel<<<dim3(NHEAD * S_SEQ), dim3(ENT_TPB), 0, stream>>>(attn, attnh);

    // out_h = attn_h @ v_h   (A = f16 attn, B = vt rows h*64..h*64+63)
    gemm_h_wmma<true, false, false><<<dim3(S_SEQ / BM, DHEAD / BN, NHEAD), blk, 0, stream>>>(
        attnh, S_SEQ, SS, vt, S_SEQ, (long long)DHEAD * S_SEQ, nullptr,
        abuf, DMODEL, DHEAD, S_SEQ, 1.0f);

    // output projection -> f32
    gemm_h_wmma<false, false, false><<<dim3(S_SEQ / BM, DMODEL / BN, 1), blk, 0, stream>>>(
        abuf, DMODEL, 0, WoT, DMODEL, 0, bo, ybuf, DMODEL, 0, DMODEL, 1.0f);

    // residual add + LayerNorm
    residual_ln_kernel<<<dim3(S_SEQ), blk, 0, stream>>>(x, ybuf, gamma, beta, out_main);
}
